// PLRNN_Step_87840671138025
// MI455X (gfx1250) — matrix operand order repaired and checked by reading, not compiled, not run on hardware
//
#include <hip/hip_runtime.h>

typedef __bf16 bf16_t;
typedef __attribute__((ext_vector_type(16))) __bf16 v16bf;
typedef __attribute__((ext_vector_type(8)))  __bf16 v8bf;
typedef __attribute__((ext_vector_type(8)))  float  v8f;
typedef __attribute__((ext_vector_type(4)))  float  v4f;

namespace {
constexpr int kB   = 8192;
constexpr int kDZ  = 1024;
constexpr int kDS  = 64;
constexpr int kDH  = 4096;
constexpr int kBM  = 64;    // batch rows per workgroup
constexpr int kKC  = 256;   // hidden-chunk width per outer step
constexpr int kTPB = 512;   // 16 x wave32
}

__device__ __forceinline__ v8f wmma_bf16f32(v16bf a, v16bf b, v8f c) {
  // D = A(16x32 bf16) x B(32x16 bf16) + C(16x16 f32)
  return __builtin_amdgcn_wmma_f32_16x16x32_bf16(false, a, false, b, (short)0, c,
                                                 false, false);
}

// A-operand fragment (16x32 bf16) from an LDS tile with 16-byte-chunk XOR
// swizzle. A layout (ISA 7.12.2): lanes 0-15 and 16-31 both carry rows
// M=0..15; half h supplies K runs {k0+8h..+7} and {k0+16+8h..+7}.
__device__ __forceinline__ v16bf ld_a_swz(const bf16_t* lds, int row, int pitch,
                                          int k0, int h, int mask) {
  int c0 = (k0 >> 3) + h;       // chunk of first 8-element K run
  int c1 = c0 + 2;              // chunk of second run (K += 16)
  v8bf lo = *(const v8bf*)(lds + row * pitch + ((c0 ^ (row & mask)) << 3));
  v8bf hi = *(const v8bf*)(lds + row * pitch + ((c1 ^ (row & mask)) << 3));
  v16bf r;
#pragma unroll
  for (int i = 0; i < 8; ++i) { r[i] = lo[i]; r[i + 8] = hi[i]; }
  return r;
}

// B-operand fragment (32x16 bf16): lane supplies column n = lane%16, which is
// one row of the row-major weight matrix; half h covers K runs {k0+8h..+7}
// and {k0+16+8h..+7} (lanes L and L+16 together consume one full 64B line).
__device__ __forceinline__ v16bf ld_b_row(const bf16_t* rowp, int k0, int h) {
  v8bf lo = *(const v8bf*)(rowp + k0 + h * 8);
  v8bf hi = *(const v8bf*)(rowp + k0 + h * 8 + 16);
  v16bf r;
#pragma unroll
  for (int i = 0; i < 8; ++i) { r[i] = lo[i]; r[i + 8] = hi[i]; }
  return r;
}

__global__ void cvt_f32_to_bf16(const float* __restrict__ src,
                                bf16_t* __restrict__ dst, int n8) {
  int i = blockIdx.x * blockDim.x + threadIdx.x;
  if (i >= n8) return;
  const v4f* s4 = (const v4f*)src + 2 * (size_t)i;
  v4f a = s4[0], b = s4[1];
  v8bf o;
  o[0] = (bf16_t)a[0]; o[1] = (bf16_t)a[1]; o[2] = (bf16_t)a[2]; o[3] = (bf16_t)a[3];
  o[4] = (bf16_t)b[0]; o[5] = (bf16_t)b[1]; o[6] = (bf16_t)b[2]; o[7] = (bf16_t)b[3];
  ((v8bf*)dst)[i] = o;
}

__global__ __launch_bounds__(kTPB)
void plrnn_step_wmma(const float* __restrict__ z,
                     const float* __restrict__ Avec,
                     const float* __restrict__ h1,
                     const float* __restrict__ h2,
                     const bf16_t* __restrict__ zb,
                     const bf16_t* __restrict__ sb,
                     const bf16_t* __restrict__ W1b,
                     const bf16_t* __restrict__ W2b,
                     const bf16_t* __restrict__ Cb,
                     float* __restrict__ out) {
  __shared__ bf16_t lds_z[kBM * kDZ];   // 128 KB  z-tile (bf16, swizzled)
  __shared__ bf16_t lds_h[kBM * kKC];   //  32 KB  Hc tile (low part reused for s)

  const int tid  = threadIdx.x;
  const int w    = tid >> 5;            // wave 0..15
  const int lane = tid & 31;
  const int h    = lane >> 4;
  const int ln   = lane & 15;
  const int row0 = blockIdx.x * kBM;
  const int nw0  = w * 64;              // this wave's 64 output columns

  // ---- stage z-tile and s-tile into LDS ------------------------------
  {
    int r = tid >> 3, c0 = tid & 7;     // 512 threads: 8 threads per row
    const bf16_t* src = zb + (size_t)(row0 + r) * kDZ;
#pragma unroll
    for (int i = 0; i < 16; ++i) {
      int c = c0 + i * 8;               // 128 chunks of 16B per row
      v8bf v = *(const v8bf*)(src + c * 8);
      *(v8bf*)(lds_z + r * kDZ + ((c ^ (r & 15)) << 3)) = v;
    }
    // s-tile: 64x64 bf16, pitch 64, mask 7, in the low part of lds_h
    v8bf v = *(const v8bf*)(sb + (size_t)(row0 + r) * kDS + c0 * 8);
    *(v8bf*)(lds_h + r * kDS + ((c0 ^ (r & 7)) << 3)) = v;
  }
  __syncthreads();

  // ---- acc init: broadcast h2, then add s @ C^T (K = 64) -------------
  v8f acc[16];                          // [mt*4 + nt] : 64 rows x 64 cols
#pragma unroll
  for (int nt = 0; nt < 4; ++nt) {
    float hv = h2[nw0 + nt * 16 + ln];
    v8f c;
#pragma unroll
    for (int j = 0; j < 8; ++j) c[j] = hv;
#pragma unroll
    for (int mt = 0; mt < 4; ++mt) acc[mt * 4 + nt] = c;
  }
#pragma unroll
  for (int kk = 0; kk < kDS; kk += 32) {
    v16bf a[4];
#pragma unroll
    for (int mt = 0; mt < 4; ++mt)
      a[mt] = ld_a_swz(lds_h, mt * 16 + ln, kDS, kk, h, 7);
#pragma unroll
    for (int nt = 0; nt < 4; ++nt) {
      const bf16_t* crow = Cb + (size_t)(nw0 + nt * 16 + ln) * kDS;
      v16bf b = ld_b_row(crow, kk, h);
#pragma unroll
      for (int mt = 0; mt < 4; ++mt)
        acc[mt * 4 + nt] = wmma_bf16f32(a[mt], b, acc[mt * 4 + nt]);
    }
  }
  __syncthreads();                      // lds_h becomes the Hc tile below

  // ---- main loop over hidden chunks ----------------------------------
  for (int kh = 0; kh < kDH; kh += kKC) {
    // Phase 1: each wave produces 16 hidden columns (all 64 rows) of
    // Hc = relu(z W1^T + h1); one W1-row B-fragment feeds 4 WMMAs.
    const int hcol = kh + w * 16;
    float hb = h1[hcol + ln];
    v8f cc[4];
#pragma unroll
    for (int mt = 0; mt < 4; ++mt)
#pragma unroll
      for (int j = 0; j < 8; ++j) cc[mt][j] = hb;
    const bf16_t* w1row = W1b + (size_t)(hcol + ln) * kDZ;
#pragma unroll 4
    for (int k0 = 0; k0 < kDZ; k0 += 32) {
      v16bf b = ld_b_row(w1row, k0, h);            // W1 row from L2
#pragma unroll
      for (int mt = 0; mt < 4; ++mt) {
        v16bf a = ld_a_swz(lds_z, mt * 16 + ln, kDZ, k0, h, 15);
        cc[mt] = wmma_bf16f32(a, b, cc[mt]);
      }
    }
    // relu + bf16 store to lds_h (pitch 256, mask 15); lane holds column
    // colc for rows m = mt*16 + 8h + j
    const int colc = w * 16 + ln;                  // 0..255
#pragma unroll
    for (int mt = 0; mt < 4; ++mt) {
#pragma unroll
      for (int j = 0; j < 8; ++j) {
        float v = cc[mt][j] > 0.f ? cc[mt][j] : 0.f;
        int m = mt * 16 + h * 8 + j;
        lds_h[m * kKC + (((colc >> 3) ^ (m & 15)) << 3) + (colc & 7)] = (bf16_t)v;
      }
    }
    __syncthreads();

    // Phase 2: acc += Hc @ W2[:, kh:kh+256]^T ; each wave owns 64 output
    // columns, so every W2 row is fetched exactly once per block.
#pragma unroll 2
    for (int kk = 0; kk < kKC; kk += 32) {
      v16bf a[4];
#pragma unroll
      for (int mt = 0; mt < 4; ++mt)
        a[mt] = ld_a_swz(lds_h, mt * 16 + ln, kKC, kk, h, 15);
#pragma unroll
      for (int nt = 0; nt < 4; ++nt) {
        const bf16_t* w2row = W2b + (size_t)(nw0 + nt * 16 + ln) * kDH + kh;
        v16bf b = ld_b_row(w2row, kk, h);
#pragma unroll
        for (int mt = 0; mt < 4; ++mt)
          acc[mt * 4 + nt] = wmma_bf16f32(a[mt], b, acc[mt * 4 + nt]);
      }
    }
    __syncthreads();
  }

  // ---- epilogue: out = acc + A * z (exact f32) -----------------------
#pragma unroll
  for (int mt = 0; mt < 4; ++mt) {
#pragma unroll
    for (int nt = 0; nt < 4; ++nt) {
      int n = nw0 + nt * 16 + ln;
      float Av = Avec[n];
      v8f c = acc[mt * 4 + nt];
#pragma unroll
      for (int j = 0; j < 8; ++j) {
        size_t off = (size_t)(row0 + mt * 16 + h * 8 + j) * kDZ + n;
        out[off] = c[j] + Av * z[off];
      }
    }
  }
}

extern "C" void kernel_launch(void* const* d_in, const int* in_sizes, int n_in,
                              void* d_out, int out_size, void* d_ws, size_t ws_size,
                              hipStream_t stream) {
  const float* z  = (const float*)d_in[0];
  const float* s  = (const float*)d_in[1];
  const float* A  = (const float*)d_in[2];
  const float* W1 = (const float*)d_in[3];
  const float* W2 = (const float*)d_in[4];
  const float* h1 = (const float*)d_in[5];
  const float* h2 = (const float*)d_in[6];
  const float* C  = (const float*)d_in[7];
  float* out = (float*)d_out;

  // d_ws layout (bytes): zb 16MB | W1b 8MB | W2b 8MB | Cb 128KB | sb 1MB
  char* ws = (char*)d_ws;
  bf16_t* zb  = (bf16_t*)(ws);
  bf16_t* W1b = (bf16_t*)(ws + (size_t)16 * 1024 * 1024);
  bf16_t* W2b = (bf16_t*)(ws + (size_t)24 * 1024 * 1024);
  bf16_t* Cb  = (bf16_t*)(ws + (size_t)32 * 1024 * 1024);
  bf16_t* sbb = (bf16_t*)(ws + (size_t)32 * 1024 * 1024 + 128 * 1024);

  auto cvt = [&](const float* src, bf16_t* dst, int n) {
    int n8 = n / 8;
    cvt_f32_to_bf16<<<(n8 + 255) / 256, 256, 0, stream>>>(src, dst, n8);
  };
  cvt(z,  zb,  kB * kDZ);
  cvt(W1, W1b, kDH * kDZ);
  cvt(W2, W2b, kDZ * kDH);
  cvt(C,  Cb,  kDZ * kDS);
  cvt(s,  sbb, kB * kDS);

  plrnn_step_wmma<<<kB / kBM, kTPB, 0, stream>>>(z, A, h1, h2, zb, sbb,
                                                 W1b, W2b, Cb, out);
}
